// Attention_80504866996432
// MI455X (gfx1250) — compile-verified, matrix operand order checked by reading, not run on hardware
//
#include <hip/hip_runtime.h>

// ---------------------------------------------------------------------------
// Causal attention block for MI455X (gfx1250):
// bf16 WMMA pipeline, f32 accumulation, TDM (tensor_load_to_lds) staging.
// ---------------------------------------------------------------------------

typedef __attribute__((ext_vector_type(16))) __bf16 v16bf;
typedef __attribute__((ext_vector_type(8)))  float  v8f;
typedef __attribute__((ext_vector_type(4)))  unsigned int u32x4;
typedef __attribute__((ext_vector_type(4)))  int i32x4;
typedef __attribute__((ext_vector_type(8)))  int i32x8;

#define N_TOK 4096
#define D_DIM 1024
#define F_DIM 1024

#define BM 128
#define BN 128
#define BK 32
#define APAD 40   // 32 elem row + 8 elem pad  (16 DWORD + 4 DWORD)  -> 80B stride
#define KPAD 72   // 64 elem row + 8 elem pad  (32 DWORD + 4 DWORD)  -> 144B stride

__device__ __forceinline__ v8f wmma_bf16(v16bf a, v16bf b, v8f c) {
  return __builtin_amdgcn_wmma_f32_16x16x32_bf16(false, a, false, b,
                                                 (short)0, c, false, false);
}

// 16x32 bf16 fragment from LDS (row-major, K contiguous, stride `ld` elems).
// CDNA5 16-bit A/B layout: lane<16: M=lane, K={0..7,16..23};
//                          lane>=16: M=lane-16, K={8..15,24..31}.
__device__ __forceinline__ v16bf ld_frag(const __bf16* base, int ld) {
  const int lane = threadIdx.x & 31;
  const __bf16* p = base + (lane & 15) * ld + (lane >> 4) * 8;
  v16bf f;
#pragma unroll
  for (int e = 0; e < 8; ++e) f[e] = p[e];
#pragma unroll
  for (int e = 0; e < 8; ++e) f[8 + e] = p[16 + e];
  return f;
}

// ---------------------------------------------------------------------------
// Tensor Data Mover: async 2-D bf16 tile (tile_w x tile_h) global -> LDS with
// hardware row padding so the LDS row stride gets the +8-element pad.
//   pad_interval_enc: DWORDs/row  = 2^(enc+1)  (tile_w*2/4 DWORDs)
//   pad_amount_enc:   pad DWORDs  = enc+1      (4 DWORDs = 8 bf16)
// D# group0: count=1 | lds_addr | global_addr | type=2
// D# group1: data_size=2B, pad fields, tensor/tile dims, dim0 stride
// Issued by one wave; completion via TENSORcnt.
// ---------------------------------------------------------------------------
__device__ __forceinline__ void tdm_load_2d(const void* gsrc, void* lds,
                                            unsigned tile_w, unsigned tile_h,
                                            unsigned row_stride,
                                            unsigned pad_interval_enc,
                                            unsigned pad_amount_enc) {
  const unsigned long long ga = (unsigned long long)(size_t)gsrc;
  u32x4 g0;
  g0[0] = 1u;                                                   // count=1
  g0[1] = (unsigned)(size_t)lds;                                // LDS byte addr
  g0[2] = (unsigned)ga;                                         // gaddr[31:0]
  g0[3] = ((unsigned)(ga >> 32) & 0x01FFFFFFu) | (2u << 30);    // [56:32]|type=2
  i32x8 g1;
  g1[0] = (int)((1u << 16) |                                    // data_size=2B
                (1u << 20) |                                    // pad_enable
                (pad_interval_enc << 22) | (pad_amount_enc << 25));
  g1[1] = (int)(row_stride << 16);                 // tensor_dim0[15:0]=stride
  g1[2] = (int)((row_stride >> 16) | 0u);          // dim0 hi | dim1 lo (dim1=4M)
  g1[3] = (int)(64u | (tile_w << 16));             // dim1 hi (=64) | tile_dim0
  g1[4] = (int)(tile_h);                           // tile_dim1 | tile_dim2=0
  g1[5] = (int)row_stride;                         // tensor_dim0_stride[31:0]
  g1[6] = 0;                                       // stride hi | dim1_stride lo
  g1[7] = 0;
  const i32x4 z4 = {0, 0, 0, 0};
#if defined(__clang_major__) && (__clang_major__ >= 23)
  const i32x8 z8 = {0, 0, 0, 0, 0, 0, 0, 0};
  __builtin_amdgcn_tensor_load_to_lds(g0, g1, z4, z4, z8, 0);
#else
  __builtin_amdgcn_tensor_load_to_lds(g0, g1, z4, z4, 0);
#endif
}

// ---------------------------------------------------------------------------
// f32 -> bf16 conversion (4 elems/thread)
// ---------------------------------------------------------------------------
__global__ void __launch_bounds__(256) cvt_f32_bf16(const float* __restrict__ in,
                                                    __bf16* __restrict__ out,
                                                    int n) {
  int i = (blockIdx.x * 256 + threadIdx.x) * 4;
  if (i + 3 < n) {
    float4 v = *(const float4*)&in[i];
    out[i + 0] = (__bf16)v.x;
    out[i + 1] = (__bf16)v.y;
    out[i + 2] = (__bf16)v.z;
    out[i + 3] = (__bf16)v.w;
  }
}

// ---------------------------------------------------------------------------
// NT GEMM: C[M,Nn] = A[M,Kd] * B[Nn,Kd]^T + bias[Nn]
// TDM double-buffered staging; 8 waves, each a 32x64 patch (2x4 WMMA tiles).
// ---------------------------------------------------------------------------
template <int F32OUT>
__global__ void __launch_bounds__(256) gemm_nt(const __bf16* __restrict__ A,
                                               const __bf16* __restrict__ B,
                                               const float* __restrict__ bias,
                                               void* __restrict__ Cout,
                                               int M, int Nn, int Kd) {
  __shared__ __bf16 As[2][BM][APAD];
  __shared__ __bf16 Bs[2][BM][APAD];

  const int tid = threadIdx.x, lane = tid & 31, w = tid >> 5;
  const int m0 = blockIdx.y * BM, n0 = blockIdx.x * BN;
  const int wr = (w >> 1) * 32, wc = (w & 1) * 64;

  const v8f vz = {};
  v8f acc[2][4];
#pragma unroll
  for (int i = 0; i < 2; ++i)
#pragma unroll
    for (int j = 0; j < 4; ++j) acc[i][j] = vz;

  if (w == 0) {
    tdm_load_2d(&A[(size_t)m0 * Kd], &As[0][0][0], BK, BM, Kd, 3, 3);
    tdm_load_2d(&B[(size_t)n0 * Kd], &Bs[0][0][0], BK, BM, Kd, 3, 3);
  }

  int cur = 0;
  for (int kk = 0; kk < Kd; kk += BK) {
    if (w == 0) __builtin_amdgcn_s_wait_tensorcnt(0);
    __syncthreads();  // tile[cur] ready; tile[cur^1] readers from last iter done
    if (w == 0 && kk + BK < Kd) {
      tdm_load_2d(&A[(size_t)m0 * Kd + kk + BK], &As[cur ^ 1][0][0], BK, BM, Kd, 3, 3);
      tdm_load_2d(&B[(size_t)n0 * Kd + kk + BK], &Bs[cur ^ 1][0][0], BK, BM, Kd, 3, 3);
    }
    v16bf af[2], bfm[4];
#pragma unroll
    for (int mt = 0; mt < 2; ++mt) af[mt] = ld_frag(&As[cur][wr + mt * 16][0], APAD);
#pragma unroll
    for (int nt = 0; nt < 4; ++nt) bfm[nt] = ld_frag(&Bs[cur][wc + nt * 16][0], APAD);
#pragma unroll
    for (int mt = 0; mt < 2; ++mt)
#pragma unroll
      for (int nt = 0; nt < 4; ++nt)
        acc[mt][nt] = wmma_bf16(af[mt], bfm[nt], acc[mt][nt]);
    cur ^= 1;
  }

  // C/D layout: VGPR r -> (M = r + 8*(lane>=16), N = lane&15)
  const int nloc = lane & 15, mbase = (lane >> 4) * 8;
#pragma unroll
  for (int mt = 0; mt < 2; ++mt)
#pragma unroll
    for (int nt = 0; nt < 4; ++nt) {
      const int gn = n0 + wc + nt * 16 + nloc;
      const float bz = bias ? bias[gn] : 0.f;
#pragma unroll
      for (int r = 0; r < 8; ++r) {
        const int gm = m0 + wr + mt * 16 + mbase + r;
        const float v = acc[mt][nt][r] + bz;
        if (F32OUT) ((float*)Cout)[(size_t)gm * Nn + gn] = v;
        else        ((__bf16*)Cout)[(size_t)gm * Nn + gn] = (__bf16)v;
      }
    }
}

// ---------------------------------------------------------------------------
// Fused causal scores + softmax: P[i,j] = softmax_j(Q[i,:]·K[j,:]*scale), j<=i
// Block = 16 query rows; Q row-block resident in LDS; K tiles via TDM with
// double buffering (prefetch stays within each jt).  Pass 0: online (max,sum)
// per wave strip + cross-wave combine.  Pass 1: recompute, write bf16 P.
// ---------------------------------------------------------------------------
#define QPAD (D_DIM + 8)   // 2064B row stride (16B multiple)

__global__ void __launch_bounds__(256) attn_scores_softmax(
    const __bf16* __restrict__ Q, const __bf16* __restrict__ Km,
    __bf16* __restrict__ P) {
  __shared__ __bf16 Qs[16][QPAD];
  __shared__ __bf16 Ks[2][128][KPAD];
  __shared__ float wm[8][16], wl[8][16];
  __shared__ float gmax[16], ginv[16];

  const int tid = threadIdx.x, lane = tid & 31, w = tid >> 5;
  const int i0 = blockIdx.x * 16;

  // Preload 16x1024 Q row-block: batch loads into regs, then LDS stores.
  {
    const int r = tid >> 4;
    const int c0 = (tid & 15) * 64;
    uint4 t[8];
#pragma unroll
    for (int q = 0; q < 8; ++q)
      t[q] = *(const uint4*)&Q[(size_t)(i0 + r) * D_DIM + c0 + q * 8];
#pragma unroll
    for (int q = 0; q < 8; ++q)
      *(uint4*)&Qs[r][c0 + q * 8] = t[q];
  }

  const int jtiles = (i0 >> 7) + 1;
  const float scale = 0.03125f;             // 1/sqrt(1024)
  const int nloc = lane & 15, mbase = (lane >> 4) * 8;

  float m_loc[8], l_loc[8], gmr[8], glr[8];
#pragma unroll
  for (int r = 0; r < 8; ++r) { m_loc[r] = -3.0e38f; l_loc[r] = 0.f; gmr[r] = 0.f; glr[r] = 0.f; }

  for (int pass = 0; pass < 2; ++pass) {
    for (int jt = 0; jt < jtiles; ++jt) {
      if (w == 0)   // prologue: K tile (jt, kk=0): 128 rows x 64 cols
        tdm_load_2d(&Km[(size_t)(jt * 128) * D_DIM], &Ks[0][0][0], 64, 128, D_DIM, 4, 3);
      int cur = 0;
      v8f acc = {};
      for (int kk = 0; kk < D_DIM; kk += 64) {
        if (w == 0) __builtin_amdgcn_s_wait_tensorcnt(0);
        __syncthreads();
        if (w == 0 && kk + 64 < D_DIM)
          tdm_load_2d(&Km[(size_t)(jt * 128) * D_DIM + kk + 64], &Ks[cur ^ 1][0][0],
                      64, 128, D_DIM, 4, 3);
#pragma unroll
        for (int c = 0; c < 2; ++c) {
          v16bf a = ld_frag(&Qs[0][kk + c * 32], QPAD);
          v16bf b = ld_frag(&Ks[cur][w * 16][c * 32], KPAD);
          acc = wmma_bf16(a, b, acc);
        }
        cur ^= 1;
      }
      const int jb = jt * 128 + w * 16 + nloc;
      if (pass == 0) {
#pragma unroll
        for (int r = 0; r < 8; ++r) {
          const int i = i0 + mbase + r;
          float s = acc[r] * scale;
          const bool valid = (jb <= i);
          s = valid ? s : -3.0e38f;
          float t = s;                      // 16-lane butterfly max (wave32)
          t = fmaxf(t, __shfl_xor(t, 1));
          t = fmaxf(t, __shfl_xor(t, 2));
          t = fmaxf(t, __shfl_xor(t, 4));
          t = fmaxf(t, __shfl_xor(t, 8));
          const float mnew = fmaxf(m_loc[r], t);
          float p = valid ? __expf(s - mnew) : 0.f;
          p += __shfl_xor(p, 1);
          p += __shfl_xor(p, 2);
          p += __shfl_xor(p, 4);
          p += __shfl_xor(p, 8);
          l_loc[r] = l_loc[r] * __expf(m_loc[r] - mnew) + p;
          m_loc[r] = mnew;
        }
      } else {
#pragma unroll
        for (int r = 0; r < 8; ++r) {
          const int i = i0 + mbase + r;
          const float s = acc[r] * scale;
          const bool valid = (jb <= i);
          const float p = valid ? __expf(s - gmr[r]) * glr[r] : 0.f;
          P[(size_t)i * N_TOK + jb] = (__bf16)p;
        }
      }
    }
    if (pass == 0) {  // combine the 8 wave-strip (max,sum) pairs
      if ((lane & 15) == 0) {
        const int hb = (lane >> 4) * 8;
#pragma unroll
        for (int r = 0; r < 8; ++r) { wm[w][hb + r] = m_loc[r]; wl[w][hb + r] = l_loc[r]; }
      }
      __syncthreads();
      if (tid < 16) {
        float m = -3.0e38f;
        for (int ww = 0; ww < 8; ++ww) m = fmaxf(m, wm[ww][tid]);
        float l = 0.f;
        for (int ww = 0; ww < 8; ++ww) l += wl[ww][tid] * __expf(wm[ww][tid] - m);
        gmax[tid] = m;
        ginv[tid] = 1.0f / l;
      }
      __syncthreads();
#pragma unroll
      for (int r = 0; r < 8; ++r) { gmr[r] = gmax[mbase + r]; glr[r] = ginv[mbase + r]; }
    }
  }
}

// ---------------------------------------------------------------------------
// Z[M,Nn] = P[M,Kd] * V[Kd,Nn]   (NN GEMM).  P tiles via TDM (double buffer);
// V tiles transposed into LDS manually (batched regs), double-buffered so the
// staging overlaps the TDM.  Causal k-limit: kend = m0 + BM.
// ---------------------------------------------------------------------------
__global__ void __launch_bounds__(256) gemm_pv(const __bf16* __restrict__ P,
                                               const __bf16* __restrict__ V,
                                               __bf16* __restrict__ Z,
                                               int M, int Nn, int Kd) {
  __shared__ __bf16 Ps[2][BM][APAD];
  __shared__ __bf16 VsT[2][BN][APAD];

  const int tid = threadIdx.x, lane = tid & 31, w = tid >> 5;
  const int m0 = blockIdx.y * BM, n0 = blockIdx.x * BN;
  const int wr = (w >> 1) * 32, wc = (w & 1) * 64;

  const v8f vz = {};
  v8f acc[2][4];
#pragma unroll
  for (int i = 0; i < 2; ++i)
#pragma unroll
    for (int j = 0; j < 4; ++j) acc[i][j] = vz;

  const int vr = tid >> 3, vc = (tid & 7) * 16;    // V tile mapping (32 x 128)
  const int kend = (Kd < m0 + BM) ? Kd : (m0 + BM);

  // Stage V tile kk: load 2x uint4 into regs, scatter transposed into LDS.
  auto stageV = [&](int buf, int kk) {
    uint4 t0 = *(const uint4*)&V[(size_t)(kk + vr) * Nn + n0 + vc];
    uint4 t1 = *(const uint4*)&V[(size_t)(kk + vr) * Nn + n0 + vc + 8];
    const __bf16* e0 = (const __bf16*)&t0;
    const __bf16* e1 = (const __bf16*)&t1;
#pragma unroll
    for (int e = 0; e < 8; ++e) {
      VsT[buf][vc + e][vr]     = e0[e];
      VsT[buf][vc + 8 + e][vr] = e1[e];
    }
  };

  if (w == 0) tdm_load_2d(&P[(size_t)m0 * Kd], &Ps[0][0][0], BK, BM, Kd, 3, 3);
  stageV(0, 0);

  int cur = 0;
  for (int kk = 0; kk < kend; kk += BK) {
    if (w == 0) __builtin_amdgcn_s_wait_tensorcnt(0);
    __syncthreads();  // Ps[cur] DMA done + VsT[cur] stores visible
    if (kk + BK < kend) {
      if (w == 0)
        tdm_load_2d(&P[(size_t)m0 * Kd + kk + BK], &Ps[cur ^ 1][0][0], BK, BM, Kd, 3, 3);
      stageV(cur ^ 1, kk + BK);
    }
    v16bf af[2], bfm[4];
#pragma unroll
    for (int mt = 0; mt < 2; ++mt) af[mt] = ld_frag(&Ps[cur][wr + mt * 16][0], APAD);
#pragma unroll
    for (int nt = 0; nt < 4; ++nt) bfm[nt] = ld_frag(&VsT[cur][wc + nt * 16][0], APAD);
#pragma unroll
    for (int mt = 0; mt < 2; ++mt)
#pragma unroll
      for (int nt = 0; nt < 4; ++nt)
        acc[mt][nt] = wmma_bf16(af[mt], bfm[nt], acc[mt][nt]);
    cur ^= 1;
  }

  const int nloc = lane & 15, mbase = (lane >> 4) * 8;
#pragma unroll
  for (int mt = 0; mt < 2; ++mt)
#pragma unroll
    for (int nt = 0; nt < 4; ++nt) {
      const int gn = n0 + wc + nt * 16 + nloc;
#pragma unroll
      for (int r = 0; r < 8; ++r) {
        const int gm = m0 + wr + mt * 16 + mbase + r;
        Z[(size_t)gm * Nn + gn] = (__bf16)acc[mt][nt][r];
      }
    }
}

// ---------------------------------------------------------------------------
// Host-side orchestration (graph-capture safe: launches only).
// Workspace layout (80 MB total):
//   xb 8M | wqb 2M | wkb 2M | wvb 2M | pwb 2M | Q 8M | K 8M | V 8M | P 32M | Z 8M
// ---------------------------------------------------------------------------
extern "C" void kernel_launch(void* const* d_in, const int* in_sizes, int n_in,
                              void* d_out, int out_size, void* d_ws, size_t ws_size,
                              hipStream_t stream) {
  const float* x    = (const float*)d_in[0];
  const float* wq_w = (const float*)d_in[1];
  const float* wq_b = (const float*)d_in[2];
  const float* wk_w = (const float*)d_in[3];
  const float* wk_b = (const float*)d_in[4];
  const float* wv_w = (const float*)d_in[5];
  const float* wv_b = (const float*)d_in[6];
  const float* pj_w = (const float*)d_in[7];
  const float* pj_b = (const float*)d_in[8];
  float* out = (float*)d_out;

  char* p = (char*)d_ws;
  auto take = [&](size_t bytes) { char* r = p; p += bytes; return r; };
  __bf16* xb  = (__bf16*)take((size_t)N_TOK * D_DIM * 2);
  __bf16* wqb = (__bf16*)take((size_t)F_DIM * D_DIM * 2);
  __bf16* wkb = (__bf16*)take((size_t)F_DIM * D_DIM * 2);
  __bf16* wvb = (__bf16*)take((size_t)F_DIM * D_DIM * 2);
  __bf16* pwb = (__bf16*)take((size_t)F_DIM * F_DIM * 2);
  __bf16* Qb  = (__bf16*)take((size_t)N_TOK * F_DIM * 2);
  __bf16* Kb  = (__bf16*)take((size_t)N_TOK * F_DIM * 2);
  __bf16* Vb  = (__bf16*)take((size_t)N_TOK * F_DIM * 2);
  __bf16* Pb  = (__bf16*)take((size_t)N_TOK * N_TOK * 2);
  __bf16* Zb  = (__bf16*)take((size_t)N_TOK * F_DIM * 2);

  const dim3 blk(256);
  // 1) f32 -> bf16 conversions
  cvt_f32_bf16<<<(N_TOK * D_DIM) / 1024, blk, 0, stream>>>(x,    xb,  N_TOK * D_DIM);
  cvt_f32_bf16<<<(F_DIM * D_DIM) / 1024, blk, 0, stream>>>(wq_w, wqb, F_DIM * D_DIM);
  cvt_f32_bf16<<<(F_DIM * D_DIM) / 1024, blk, 0, stream>>>(wk_w, wkb, F_DIM * D_DIM);
  cvt_f32_bf16<<<(F_DIM * D_DIM) / 1024, blk, 0, stream>>>(wv_w, wvb, F_DIM * D_DIM);
  cvt_f32_bf16<<<(F_DIM * F_DIM) / 1024, blk, 0, stream>>>(pj_w, pwb, F_DIM * F_DIM);

  // 2) QKV projections (bf16 out)
  const dim3 gQKV(F_DIM / BN, N_TOK / BM);
  gemm_nt<0><<<gQKV, blk, 0, stream>>>(xb, wqb, wq_b, Qb, N_TOK, F_DIM, D_DIM);
  gemm_nt<0><<<gQKV, blk, 0, stream>>>(xb, wkb, wk_b, Kb, N_TOK, F_DIM, D_DIM);
  gemm_nt<0><<<gQKV, blk, 0, stream>>>(xb, wvb, wv_b, Vb, N_TOK, F_DIM, D_DIM);

  // 3) causal scores + softmax -> P (bf16)
  attn_scores_softmax<<<N_TOK / 16, blk, 0, stream>>>(Qb, Kb, Pb);

  // 4) Z = P * V (bf16 out, causal k-limit)
  gemm_pv<<<dim3(F_DIM / BN, N_TOK / BM), blk, 0, stream>>>(Pb, Vb, Zb, N_TOK, F_DIM, N_TOK);

  // 5) out = Z * proj_w^T + proj_b (f32 out)
  gemm_nt<1><<<gQKV, blk, 0, stream>>>(Zb, pwb, pj_b, out, N_TOK, F_DIM, F_DIM);
}
